// Desc_Seq2Seq_91053306675696
// MI455X (gfx1250) — compile-verified, matrix operand order checked by reading, not run on hardware
//
#include <hip/hip_runtime.h>
#include <hip/hip_bf16.h>

// Problem constants (match reference)
#define B_   512   // batch
#define T_   512   // encoder sequence length
#define F_   128   // encoder feature size
#define H_   512   // hidden size
#define G4_  2048  // 4*H
#define P_   96    // pred len

// GEMM tiling
#define BM_  64    // block M tile
#define BN_  128   // block N tile
#define LDSS 40    // LDS row stride in elements (80B: pads 64B rows to avoid bank aliasing)

typedef __attribute__((ext_vector_type(16))) __bf16 v16bf;
typedef __attribute__((ext_vector_type(8)))  __bf16 v8bf;
typedef __attribute__((ext_vector_type(8)))  float  v8f;
typedef __attribute__((ext_vector_type(4)))  int    v4i;

// Address-space-qualified v4i for the async-to-LDS builtin
typedef __attribute__((address_space(1))) v4i gv4i;  // global
typedef __attribute__((address_space(3))) v4i lv4i;  // LDS

#if defined(__has_builtin)
# if __has_builtin(__builtin_amdgcn_global_load_async_to_lds_b128)
#  define ASYNC_LDS 1
# endif
#endif
#ifndef ASYNC_LDS
# define ASYNC_LDS 0
#endif

// ---------------------------------------------------------------------------
// WMMA fragment load: 16x32 bf16 tile from row-major storage (global or LDS).
// Lane l<16: row M=l, K = {0..7, 16..23}; lane l>=16: row M=l-16, K = {8..15, 24..31}.
// Per lane: two contiguous 16-byte chunks at (k0 + half*8) and (k0 + half*8 + 16).
// ---------------------------------------------------------------------------
__device__ __forceinline__ v16bf frag_load(const __bf16* p) {
  v8bf lo = *(const v8bf*)(p);
  v8bf hi = *(const v8bf*)(p + 16);
  return __builtin_shufflevector(lo, hi, 0,1,2,3,4,5,6,7,8,9,10,11,12,13,14,15);
}

__device__ __forceinline__ v8f wmma_bf16(v16bf a, v16bf b, v8f c) {
  return __builtin_amdgcn_wmma_f32_16x16x32_bf16(
      false, a, false, b, (short)0, c, false, false);
}

// Stage one 16-byte chunk global -> LDS (async path when available).
// Global side: generic and global addresses are value-identical (ptrtoint route).
// LDS side: per the ISA aperture rule the low 32 bits of the
// generic pointer are the LDS offset -> valid addrspace(3) pointer.
__device__ __forceinline__ void stage16(const __bf16* g, __bf16* l) {
#if ASYNC_LDS
  gv4i* gp = (gv4i*)(unsigned long long)(uintptr_t)g;
  lv4i* lp = (lv4i*)(unsigned int)(uintptr_t)l;
  __builtin_amdgcn_global_load_async_to_lds_b128(gp, lp, 0, 0);
#else
  *(v8bf*)l = *(const v8bf*)g;
#endif
}

__device__ __forceinline__ void async_wait() {
#if ASYNC_LDS
# if __has_builtin(__builtin_amdgcn_s_wait_asynccnt)
  __builtin_amdgcn_s_wait_asynccnt(0);
# else
  asm volatile("s_wait_asynccnt 0x0" ::: "memory");
# endif
#endif
}

// ---------------------------------------------------------------------------
// gates[B,4H] = A1[B,K1] @ W1[4H,K1]^T + A2[B,K2] @ W2[4H,K2]^T   (bias in cell)
// Block = 256 threads (8 waves, 2M x 4N); each wave computes a 32x32 tile
// (2 A-frags x 2 B-frags -> 4 WMMAs per k-step, register reuse on both operands).
// A panel (64 rows x 32 K) is staged into LDS per k-step via async-to-LDS;
// W comes from global (bf16 weights are L2-resident across all time steps).
// Grid = (G4/128, B/64). K1,K2 multiples of 32.
// ---------------------------------------------------------------------------
__global__ __launch_bounds__(256) void gates_gemm_wmma(
    const __bf16* __restrict__ A1, long lda1, const __bf16* __restrict__ W1, int K1,
    const __bf16* __restrict__ A2, long lda2, const __bf16* __restrict__ W2, int K2,
    float* __restrict__ gates)
{
  __shared__ __align__(16) __bf16 sA[BM_ * LDSS];

  const int tid  = threadIdx.x;
  const int lane = tid & 31;
  const int wave = tid >> 5;
  const int row  = lane & 15;
  const int half = lane >> 4;
  const int wm   = wave & 1;         // 2 wave rows (M)
  const int wn   = wave >> 1;        // 4 wave cols (N)
  const int bm0  = blockIdx.y * BM_;
  const int tm   = bm0 + wm * 32;
  const int tn   = blockIdx.x * BN_ + wn * 32;

  // staging assignment: 256 threads copy 64 rows x 4 chunks of 16B
  const int crow = tid >> 2;         // 0..63
  const int cchk = (tid & 3) * 8;    // element offset 0,8,16,24

  v8f acc00 = {}, acc01 = {}, acc10 = {}, acc11 = {};

  #pragma unroll
  for (int seg = 0; seg < 2; ++seg) {
    const __bf16* A  = seg ? A2 : A1;
    const __bf16* W  = seg ? W2 : W1;
    const long   lda = seg ? lda2 : lda1;
    const int    K   = seg ? K2  : K1;

    for (int k = 0; k < K; k += 32) {
      __syncthreads();   // protect LDS panel still being read
      stage16(A + (long)(bm0 + crow) * lda + k + cchk,
              &sA[crow * LDSS + cchk]);
      async_wait();
      __syncthreads();   // panel visible to all waves

      // A fragments from LDS (shared across the 4 N-waves)
      v16bf a0 = frag_load(&sA[(wm * 32 + row) * LDSS + half * 8]);
      v16bf a1 = frag_load(&sA[(wm * 32 + 16 + row) * LDSS + half * 8]);

      // B fragments from global (row-major W[4H][K]; B[k][n] = W[n][k])
      const __bf16* wp = W + (long)(tn + row) * (long)K + k + half * 8;
      __builtin_prefetch(wp + 32, 0, 0);  // global_prefetch_b8 (L2-resident)
      v16bf b0 = frag_load(wp);
      v16bf b1 = frag_load(wp + (long)16 * K);

      acc00 = wmma_bf16(a0, b0, acc00);
      acc01 = wmma_bf16(a0, b1, acc01);
      acc10 = wmma_bf16(a1, b0, acc10);
      acc11 = wmma_bf16(a1, b1, acc11);
    }
  }

  // C/D layout: VGPR r -> M = half*8 + r, N = lane&15
  {
    float* o = gates + (long)(tm + half * 8) * G4_ + tn + row;
    #pragma unroll
    for (int r = 0; r < 8; ++r) {
      o[(long)r * G4_]           = acc00[r];
      o[(long)r * G4_ + 16]      = acc01[r];
      o[(long)(r + 16) * G4_]      = acc10[r];
      o[(long)(r + 16) * G4_ + 16] = acc11[r];
    }
  }
}

// ---------------------------------------------------------------------------
// Fused LSTM cell pointwise: gates + biases -> activations -> (c, h_bf16)
// ---------------------------------------------------------------------------
__global__ __launch_bounds__(256) void lstm_cell_kernel(
    const float* __restrict__ gates,
    const float* __restrict__ b_ih, const float* __restrict__ b_hh,
    float* __restrict__ c, __bf16* __restrict__ h_bf)
{
  int idx = blockIdx.x * blockDim.x + threadIdx.x;
  if (idx >= B_ * H_) return;
  int b = idx >> 9;        // /H_
  int j = idx & (H_ - 1);  // %H_
  const float* g = gates + (long)b * G4_;
  float gi = g[j]          + b_ih[j]          + b_hh[j];
  float gf = g[j + H_]     + b_ih[j + H_]     + b_hh[j + H_];
  float gg = g[j + 2*H_]   + b_ih[j + 2*H_]   + b_hh[j + 2*H_];
  float go = g[j + 3*H_]   + b_ih[j + 3*H_]   + b_hh[j + 3*H_];
  float i_ = 1.0f / (1.0f + __expf(-gi));
  float f_ = 1.0f / (1.0f + __expf(-gf));
  float g_ = tanhf(gg);
  float o_ = 1.0f / (1.0f + __expf(-go));
  float cn = f_ * c[idx] + i_ * g_;
  c[idx] = cn;
  h_bf[idx] = (__bf16)(o_ * tanhf(cn));
}

// ---------------------------------------------------------------------------
// Utilities
// ---------------------------------------------------------------------------
__global__ void f32_to_bf16_kernel(const float* __restrict__ in,
                                   __bf16* __restrict__ out, long n) {
  long i = (long)blockIdx.x * blockDim.x + threadIdx.x;
  if (i < n) out[i] = (__bf16)in[i];
}

__global__ void zero_f32_kernel(float* p, long n) {
  long i = (long)blockIdx.x * blockDim.x + threadIdx.x;
  if (i < n) p[i] = 0.0f;
}

__global__ void zero_bf16_kernel(__bf16* p, long n) {
  long i = (long)blockIdx.x * blockDim.x + threadIdx.x;
  if (i < n) p[i] = (__bf16)0.0f;
}

// dec_W_ih0 is (4H x 1); pad to (4H x 32) bf16, col 0 = weight, rest 0
__global__ void pad_dec_wih0_kernel(const float* __restrict__ w,
                                    __bf16* __restrict__ out) {
  int i = blockIdx.x * blockDim.x + threadIdx.x;
  if (i >= G4_ * 32) return;
  int n = i >> 5, k = i & 31;
  out[i] = (k == 0) ? (__bf16)w[n] : (__bf16)0.0f;
}

// dec initial input: sum(X_decode[b,:,:]) -> xpad[b*32]; one wave per batch row
__global__ __launch_bounds__(256) void dec_init_input_kernel(
    const float* __restrict__ Xd, __bf16* __restrict__ xpad) {
  int gw = (blockIdx.x * 256 + threadIdx.x) >> 5;
  int lane = threadIdx.x & 31;
  if (gw >= B_) return;
  const float* p = Xd + (long)gw * (P_ * 8);
  float s = 0.0f;
  for (int i = lane; i < P_ * 8; i += 32) s += p[i];
  #pragma unroll
  for (int o = 16; o > 0; o >>= 1) s += __shfl_xor(s, o);
  if (lane == 0) xpad[gw * 32] = (__bf16)s;
}

// fc projection + teacher-forcing feedback; one wave per batch row
__global__ __launch_bounds__(256) void fc_step_kernel(
    const __bf16* __restrict__ h_top, const float* __restrict__ fcW,
    const float* __restrict__ fcB, const float* __restrict__ y,
    const int* __restrict__ tf_mask, int t,
    float* __restrict__ out, __bf16* __restrict__ xpad)
{
  int gw = (blockIdx.x * 256 + threadIdx.x) >> 5;
  int lane = threadIdx.x & 31;
  if (gw >= B_) return;
  const __bf16* h = h_top + (long)gw * H_;
  float s = 0.0f;
  for (int k = lane; k < H_; k += 32) s += (float)h[k] * fcW[k];
  #pragma unroll
  for (int o = 16; o > 0; o >>= 1) s += __shfl_xor(s, o);
  if (lane == 0) {
    float o_ = s + fcB[0];
    out[(long)gw * P_ + t] = o_;
    bool tf = (tf_mask[t] != 0) && (t < P_ - 1);
    float ynext = (t < P_ - 1) ? y[(long)gw * P_ + (t + 1)] : 0.0f;
    xpad[gw * 32] = (__bf16)(tf ? ynext : o_);
  }
}

// ---------------------------------------------------------------------------
// Host orchestration
// ---------------------------------------------------------------------------
static inline dim3 grid1d(long n, int blk) { return dim3((unsigned)((n + blk - 1) / blk)); }

extern "C" void kernel_launch(void* const* d_in, const int* in_sizes, int n_in,
                              void* d_out, int out_size, void* d_ws, size_t ws_size,
                              hipStream_t stream) {
  const float* X_encode = (const float*)d_in[0];
  const float* X_decode = (const float*)d_in[1];
  const float* y        = (const float*)d_in[2];
  const int*   tf_mask  = (const int*)d_in[3];
  const float* encWih0 = (const float*)d_in[4];
  const float* encWhh0 = (const float*)d_in[5];
  const float* encBih0 = (const float*)d_in[6];
  const float* encBhh0 = (const float*)d_in[7];
  const float* encWih1 = (const float*)d_in[8];
  const float* encWhh1 = (const float*)d_in[9];
  const float* encBih1 = (const float*)d_in[10];
  const float* encBhh1 = (const float*)d_in[11];
  const float* decWih0 = (const float*)d_in[12];
  const float* decWhh0 = (const float*)d_in[13];
  const float* decBih0 = (const float*)d_in[14];
  const float* decBhh0 = (const float*)d_in[15];
  const float* decWih1 = (const float*)d_in[16];
  const float* decWhh1 = (const float*)d_in[17];
  const float* decBih1 = (const float*)d_in[18];
  const float* decBhh1 = (const float*)d_in[19];
  const float* fcW     = (const float*)d_in[20];
  const float* fcB     = (const float*)d_in[21];
  float* out = (float*)d_out;

  // Carve workspace (256B aligned)
  char* ws = (char*)d_ws;
  size_t off = 0;
  auto carve = [&](size_t bytes) -> void* {
    void* p = ws + off;
    off = (off + bytes + 255) & ~(size_t)255;
    return p;
  };
  __bf16* Xbf       = (__bf16*)carve((size_t)B_ * T_ * F_ * 2);
  __bf16* eWih0     = (__bf16*)carve((size_t)G4_ * F_ * 2);
  __bf16* eWhh0     = (__bf16*)carve((size_t)G4_ * H_ * 2);
  __bf16* eWih1     = (__bf16*)carve((size_t)G4_ * H_ * 2);
  __bf16* eWhh1     = (__bf16*)carve((size_t)G4_ * H_ * 2);
  __bf16* dWih0pad  = (__bf16*)carve((size_t)G4_ * 32 * 2);
  __bf16* dWhh0     = (__bf16*)carve((size_t)G4_ * H_ * 2);
  __bf16* dWih1     = (__bf16*)carve((size_t)G4_ * H_ * 2);
  __bf16* dWhh1     = (__bf16*)carve((size_t)G4_ * H_ * 2);
  float*  gates     = (float*)carve((size_t)B_ * G4_ * 4);
  float*  c0        = (float*)carve((size_t)B_ * H_ * 4);
  float*  c1        = (float*)carve((size_t)B_ * H_ * 4);
  __bf16* h0bf      = (__bf16*)carve((size_t)B_ * H_ * 2);
  __bf16* h1bf      = (__bf16*)carve((size_t)B_ * H_ * 2);
  __bf16* xpad      = (__bf16*)carve((size_t)B_ * 32 * 2);

  const int blk = 256;

  // --- precision conversion (weights L2-resident in bf16 afterwards) ---
  f32_to_bf16_kernel<<<grid1d((long)B_*T_*F_, blk), blk, 0, stream>>>(X_encode, Xbf, (long)B_*T_*F_);
  f32_to_bf16_kernel<<<grid1d((long)G4_*F_, blk), blk, 0, stream>>>(encWih0, eWih0, (long)G4_*F_);
  f32_to_bf16_kernel<<<grid1d((long)G4_*H_, blk), blk, 0, stream>>>(encWhh0, eWhh0, (long)G4_*H_);
  f32_to_bf16_kernel<<<grid1d((long)G4_*H_, blk), blk, 0, stream>>>(encWih1, eWih1, (long)G4_*H_);
  f32_to_bf16_kernel<<<grid1d((long)G4_*H_, blk), blk, 0, stream>>>(encWhh1, eWhh1, (long)G4_*H_);
  f32_to_bf16_kernel<<<grid1d((long)G4_*H_, blk), blk, 0, stream>>>(decWhh0, dWhh0, (long)G4_*H_);
  f32_to_bf16_kernel<<<grid1d((long)G4_*H_, blk), blk, 0, stream>>>(decWih1, dWih1, (long)G4_*H_);
  f32_to_bf16_kernel<<<grid1d((long)G4_*H_, blk), blk, 0, stream>>>(decWhh1, dWhh1, (long)G4_*H_);
  pad_dec_wih0_kernel<<<grid1d((long)G4_*32, blk), blk, 0, stream>>>(decWih0, dWih0pad);

  // --- zero initial state ---
  zero_f32_kernel<<<grid1d((long)B_*H_, blk), blk, 0, stream>>>(c0, (long)B_*H_);
  zero_f32_kernel<<<grid1d((long)B_*H_, blk), blk, 0, stream>>>(c1, (long)B_*H_);
  zero_bf16_kernel<<<grid1d((long)B_*H_, blk), blk, 0, stream>>>(h0bf, (long)B_*H_);
  zero_bf16_kernel<<<grid1d((long)B_*H_, blk), blk, 0, stream>>>(h1bf, (long)B_*H_);
  zero_bf16_kernel<<<grid1d((long)B_*32, blk), blk, 0, stream>>>(xpad, (long)B_*32);

  dim3 ggrid(G4_ / BN_, B_ / BM_);   // (16, 8)
  dim3 cgrid((B_ * H_) / blk);

  // --- encoder: 512 steps x 2 layers ---
  for (int t = 0; t < T_; ++t) {
    gates_gemm_wmma<<<ggrid, blk, 0, stream>>>(
        Xbf + (long)t * F_, (long)T_ * F_, eWih0, F_,
        h0bf, (long)H_, eWhh0, H_, gates);
    lstm_cell_kernel<<<cgrid, blk, 0, stream>>>(gates, encBih0, encBhh0, c0, h0bf);
    gates_gemm_wmma<<<ggrid, blk, 0, stream>>>(
        h0bf, (long)H_, eWih1, H_,
        h1bf, (long)H_, eWhh1, H_, gates);
    lstm_cell_kernel<<<cgrid, blk, 0, stream>>>(gates, encBih1, encBhh1, c1, h1bf);
  }

  // --- decoder initial input ---
  dec_init_input_kernel<<<grid1d((long)B_ * 32, blk), blk, 0, stream>>>(X_decode, xpad);

  // --- decoder: 96 steps x 2 layers + fc/feedback ---
  for (int t = 0; t < P_; ++t) {
    gates_gemm_wmma<<<ggrid, blk, 0, stream>>>(
        xpad, (long)32, dWih0pad, 32,
        h0bf, (long)H_, dWhh0, H_, gates);
    lstm_cell_kernel<<<cgrid, blk, 0, stream>>>(gates, decBih0, decBhh0, c0, h0bf);
    gates_gemm_wmma<<<ggrid, blk, 0, stream>>>(
        h0bf, (long)H_, dWih1, H_,
        h1bf, (long)H_, dWhh1, H_, gates);
    lstm_cell_kernel<<<cgrid, blk, 0, stream>>>(gates, decBih1, decBhh1, c1, h1bf);
    fc_step_kernel<<<grid1d((long)B_ * 32, blk), blk, 0, stream>>>(
        h1bf, fcW, fcB, y, tf_mask, t, out, xpad);
  }
}
